// Gemma2DecoderLayer_18141941858821
// MI455X (gfx1250) — matrix operand order repaired and checked by reading, not compiled, not run on hardware
//
#include <hip/hip_runtime.h>
#include <hip/hip_fp16.h>

typedef __attribute__((ext_vector_type(16))) _Float16 v16h;
typedef __attribute__((ext_vector_type(8)))  float    v8f;
typedef __attribute__((ext_vector_type(4)))  unsigned int u32x4;
typedef __attribute__((ext_vector_type(8)))  int      i32x8;
typedef __attribute__((ext_vector_type(4)))  int      i32x4;

#define HID    3584
#define NH     16
#define NKV    8
#define DH     256
#define INTER  14336
#define WINDOW 1024
#define LOGIT_CAP 50.0f
#define ATT_SCALE 0.0625f        /* 256^-0.5 */
#define RMS_EPS   1e-6f
#define BB 2
#define SS 2048
#define TT (BB*SS)               /* 4096 tokens */

#define BM 128
#define BN 128
#define BK 32
#define BROW 129                 /* fp32 LDS row stride (TDM pad: 128DW + 1DW) */
#define AROW 40                  /* f16 LDS row stride for A tiles (padded)   */
#define KROW 264                 /* f16 LDS row stride for attn K tiles       */

#if defined(__has_builtin)
#  if __has_builtin(__builtin_amdgcn_tensor_load_to_lds)
#    define HAS_TDM 1
#  endif
#endif
#ifndef HAS_TDM
#  define HAS_TDM 0
#endif
#if __has_include(<hip/amd_detail/amd_gfx1250_TDM.h>)
#  define TDM_SIX_ARGS 1
#else
#  define TDM_SIX_ARGS 0
#endif

/* ---------------- CDNA5 helpers ---------------- */

__device__ __forceinline__ v8f wmma32(v16h a, v16h b, v8f c) {
  return __builtin_amdgcn_wmma_f32_16x16x32_f16(false, a, false, b,
                                                (short)0, c, false, false);
}

__device__ __forceinline__ v16h ld_frag(const _Float16* p) {
  union { uint4 u[2]; v16h v; } r;
  const uint4* q = (const uint4*)p;
  r.u[0] = q[0]; r.u[1] = q[1];
  return r.v;
}

__device__ __forceinline__ v16h ld_frag2(const _Float16* p0, const _Float16* p1) {
  union { uint4 u[2]; v16h v; } r;
  r.u[0] = *(const uint4*)p0;
  r.u[1] = *(const uint4*)p1;
  return r.v;
}

__device__ __forceinline__ v8f zero8() {
  v8f z = {0.f,0.f,0.f,0.f,0.f,0.f,0.f,0.f};
  return z;
}

__device__ __forceinline__ unsigned lds_off32(const void* p) {
  return (unsigned)(unsigned long long)p;   /* LDS byte offset */
}

/* async global->LDS 16B copy, per-lane scatter; tracked by ASYNCcnt */
__device__ __forceinline__ void async_cp16(unsigned lds_byte, const void* g) {
  asm volatile("global_load_async_to_lds_b128 %0, %1, off"
               :: "v"(lds_byte), "v"((unsigned long long)g)
               : "memory");
}
__device__ __forceinline__ void async_wait0() {
  asm volatile("s_wait_asynccnt 0" ::: "memory");
}

#if HAS_TDM
/* TDM: 2D fp32 tile (tile_cols x tile_rows), global row stride
 * `stride_elems`, into LDS with 1-DWORD pad per 128-DWORD row (BROW=129).
 * D# group0/group1 packing per CDNA5 ISA 8.3/8.4. */
__device__ __forceinline__ void tdm_load_tile_f32(const float* g, unsigned lds,
                                                  int tile_cols, int tile_rows,
                                                  unsigned long long stride_elems) {
  unsigned long long ga = (unsigned long long)(const void*)g;
  u32x4 g0;
  g0[0] = 1u;                                  /* count=1, user descriptor */
  g0[1] = lds;                                 /* lds_addr (bytes)         */
  g0[2] = (unsigned)(ga & 0xffffffffu);
  g0[3] = (unsigned)((ga >> 32) & 0x01ffffffu) | (2u << 30); /* type=2    */
  unsigned d0 = (unsigned)stride_elems;        /* tensor_dim0              */
  unsigned d1 = 1u << 30;                      /* tensor_dim1 (large)      */
  unsigned slo = (unsigned)(stride_elems & 0xffffffffu);
  unsigned shi = (unsigned)((stride_elems >> 32) & 0xffffu);
  i32x8 g1;
  g1[0] = (int)((2u << 16) | (1u << 20) | (6u << 22)); /* 4B, pad 128DW+1 */
  g1[1] = (int)((d0 & 0xffffu) << 16);
  g1[2] = (int)(((d0 >> 16) & 0xffffu) | ((d1 & 0xffffu) << 16));
  g1[3] = (int)(((d1 >> 16) & 0xffffu) | ((unsigned)tile_cols << 16));
  g1[4] = (int)((unsigned)tile_rows & 0xffffu);
  g1[5] = (int)slo;
  g1[6] = (int)shi;
  g1[7] = 0;
  i32x4 z4 = {0,0,0,0};
#if TDM_SIX_ARGS
  i32x8 z8 = {0,0,0,0,0,0,0,0};
  __builtin_amdgcn_tensor_load_to_lds(g0, g1, z4, z4, z8, 0);
#else
  __builtin_amdgcn_tensor_load_to_lds(g0, g1, z4, z4, 0);
#endif
}
#endif

/* ---------------- RMSNorm kernels ---------------- */

__global__ __launch_bounds__(256)
void k_rms_f16(const float* __restrict__ x, const float* __restrict__ w,
               _Float16* __restrict__ out) {
  __shared__ float red[256];
  size_t row = blockIdx.x;
  const float* xr = x + row * HID;
  float s = 0.f;
  for (int i = threadIdx.x; i < HID; i += 256) { float v = xr[i]; s += v * v; }
  red[threadIdx.x] = s; __syncthreads();
  for (int off = 128; off; off >>= 1) {
    if (threadIdx.x < off) red[threadIdx.x] += red[threadIdx.x + off];
    __syncthreads();
  }
  float scale = rsqrtf(red[0] * (1.0f / HID) + RMS_EPS);
  for (int i = threadIdx.x; i < HID; i += 256)
    out[row * HID + i] = (_Float16)(xr[i] * scale * (1.0f + w[i]));
}

__global__ __launch_bounds__(256)
void k_rms_addres(const float* __restrict__ x, const float* __restrict__ w,
                  const float* __restrict__ resid, float* __restrict__ out) {
  __shared__ float red[256];
  size_t row = blockIdx.x;
  const float* xr = x + row * HID;
  const float* rr = resid + row * HID;
  float s = 0.f;
  for (int i = threadIdx.x; i < HID; i += 256) { float v = xr[i]; s += v * v; }
  red[threadIdx.x] = s; __syncthreads();
  for (int off = 128; off; off >>= 1) {
    if (threadIdx.x < off) red[threadIdx.x] += red[threadIdx.x + off];
    __syncthreads();
  }
  float scale = rsqrtf(red[0] * (1.0f / HID) + RMS_EPS);
  for (int i = threadIdx.x; i < HID; i += 256)
    out[row * HID + i] = rr[i] + xr[i] * scale * (1.0f + w[i]);
}

__global__ __launch_bounds__(256)
void k_rms_f32(const float* __restrict__ x, const float* __restrict__ w,
               float* __restrict__ out) {
  __shared__ float red[256];
  size_t row = blockIdx.x;
  const float* xr = x + row * HID;
  float s = 0.f;
  for (int i = threadIdx.x; i < HID; i += 256) { float v = xr[i]; s += v * v; }
  red[threadIdx.x] = s; __syncthreads();
  for (int off = 128; off; off >>= 1) {
    if (threadIdx.x < off) red[threadIdx.x] += red[threadIdx.x + off];
    __syncthreads();
  }
  float scale = rsqrtf(red[0] * (1.0f / HID) + RMS_EPS);
  for (int i = threadIdx.x; i < HID; i += 256)
    out[row * HID + i] = xr[i] * scale * (1.0f + w[i]);
}

/* -------- Pipelined WMMA GEMM: C[M,N] = A[M,K](f16) * B[K,N](f32) --------
 * Double-buffered: TDM fetches B tile kt+1 and async-to-LDS fetches A tile
 * kt+1 while tile kt is converted + WMMA'd. 2 barriers / k-step. */

__global__ __launch_bounds__(256)
void k_gemm(const _Float16* __restrict__ A, const float* __restrict__ Bw,
            _Float16* __restrict__ Ch, float* __restrict__ Cf,
            int N, int K) {
  __shared__ _Float16 Asm[2][BM * AROW];  /* 2 x 10KB   */
  __shared__ float    Bf [2][BK * BROW];  /* 2 x 16.5KB */
  __shared__ _Float16 Bsm[BK * BN];       /* 8KB        */

  const int tid = threadIdx.x;
  const int lane = tid & 31;
  const int w = tid >> 5;
  const int wr = w >> 1, wc = w & 1;
  const int rloc = lane & 15;
  const bool hi = lane >= 16;
  const int m0 = blockIdx.x * BM;
  const int n0 = blockIdx.y * BN;

  v8f acc[2][4];
  #pragma unroll
  for (int i = 0; i < 2; ++i)
    #pragma unroll
    for (int j = 0; j < 4; ++j) acc[i][j] = zero8();

  const int bcol = tid >> 1;
  const int bkh  = tid & 1;
  const int bnt  = bcol >> 4, bcl = bcol & 15;
  const int boff = bnt * 512 + (bcl + (bkh << 4)) * 16;

  const unsigned AsmB = lds_off32(&Asm[0][0]);
  const unsigned BfB  = lds_off32(&Bf[0][0]);
  (void)BfB;

  auto issue_tile = [&](int kt2, int buf2) {
    const int k0 = kt2 * BK;
#if HAS_TDM
    if (w == 0)
      tdm_load_tile_f32(Bw + (size_t)k0 * N + n0,
                        BfB + (unsigned)buf2 * (BK * BROW * 4), BN, BK,
                        (unsigned long long)N);
#endif
    #pragma unroll
    for (int i = 0; i < 2; ++i) {
      const int c = tid * 2 + i;
      const int row = c >> 2, sub = c & 3;
      const _Float16* g = A + (size_t)(m0 + row) * K + k0 + sub * 8;
      async_cp16(AsmB + (unsigned)buf2 * (BM * AROW * 2) +
                 (unsigned)(row * AROW + sub * 8) * 2, g);
    }
  };

  const int nK = K / BK;
  issue_tile(0, 0);

  for (int kt = 0; kt < nK; ++kt) {
    const int buf = kt & 1;

    /* tile kt arrival (issued last iteration) */
    async_wait0();
#if HAS_TDM
    if (w == 0) __builtin_amdgcn_s_wait_tensorcnt(0);
#else
    for (int id = tid; id < BK * BN; id += 256) {
      const int r2 = id >> 7, c2 = id & 127;
      Bf[buf][r2 * BROW + c2] = Bw[(size_t)(kt * BK + r2) * N + n0 + c2];
    }
#endif
    __syncthreads();   /* tile kt visible; all waves done with compute kt-1 */

    if (kt + 1 < nK) issue_tile(kt + 1, buf ^ 1);

    /* convert B tile kt to frag-major f16 */
    {
      union { uint4 u[2]; _Float16 h[16]; } bf2;
      #pragma unroll
      for (int j = 0; j < 16; ++j)
        bf2.h[j] = (_Float16)Bf[buf][(bkh * 16 + j) * BROW + bcol];
      *(uint4*)&Bsm[boff]     = bf2.u[0];
      *(uint4*)&Bsm[boff + 8] = bf2.u[1];
    }
    __syncthreads();

    const int eo = hi ? 16 : 0;
    v16h af[2], bfr[4];
    #pragma unroll
    for (int i = 0; i < 2; ++i) {
      const _Float16* pr = &Asm[buf][(wr * 32 + i * 16 + rloc) * AROW + eo];
      af[i] = ld_frag2(pr, pr + 8);
    }
    #pragma unroll
    for (int j = 0; j < 4; ++j)
      bfr[j] = ld_frag(&Bsm[(wc * 4 + j) * 512 + lane * 16]);
    #pragma unroll
    for (int i = 0; i < 2; ++i)
      #pragma unroll
      for (int j = 0; j < 4; ++j)
        acc[i][j] = wmma32(af[i], bfr[j], acc[i][j]);
  }

  const int rbase = m0 + wr * 32 + (hi ? 8 : 0);
  const int cbase = n0 + wc * 64 + rloc;
  #pragma unroll
  for (int i = 0; i < 2; ++i)
    #pragma unroll
    for (int j = 0; j < 4; ++j)
      #pragma unroll
      for (int r = 0; r < 8; ++r) {
        const size_t row = rbase + i * 16 + r;
        const size_t col = cbase + j * 16;
        if (Ch) Ch[row * N + col] = (_Float16)acc[i][j][r];
        else    Cf[row * N + col] = acc[i][j][r];
      }
}

/* ------------- fused gate/up GEMM: inter = up * gelu(gate) --------------- */

__global__ __launch_bounds__(256)
void k_gateup(const _Float16* __restrict__ A, const float* __restrict__ Bg,
              const float* __restrict__ Bu, _Float16* __restrict__ Ch,
              int N, int K) {
  __shared__ _Float16 Asm[BM * AROW];
  __shared__ float    Bfg[BK * BROW];
  __shared__ float    Bfu[BK * BROW];
  __shared__ _Float16 Gsm[BK * BN];
  __shared__ _Float16 Usm[BK * BN];

  const int tid = threadIdx.x;
  const int lane = tid & 31;
  const int w = tid >> 5;
  const int wr = w >> 1, wc = w & 1;
  const int rloc = lane & 15;
  const bool hi = lane >= 16;
  const int m0 = blockIdx.x * BM;
  const int n0 = blockIdx.y * BN;

  v8f ag[2][4], au[2][4];
  #pragma unroll
  for (int i = 0; i < 2; ++i)
    #pragma unroll
    for (int j = 0; j < 4; ++j) { ag[i][j] = zero8(); au[i][j] = zero8(); }

  const int bcol = tid >> 1;
  const int bkh  = tid & 1;
  const int bnt  = bcol >> 4, bcl = bcol & 15;
  const int boff = bnt * 512 + (bcl + (bkh << 4)) * 16;

  const unsigned AsmB = lds_off32(&Asm[0]);
  const unsigned BgB  = lds_off32(&Bfg[0]);
  const unsigned BuB  = lds_off32(&Bfu[0]);
  (void)BgB; (void)BuB;

  const int nK = K / BK;
  for (int kt = 0; kt < nK; ++kt) {
    const int k0 = kt * BK;
#if HAS_TDM
    if (w == 0) {
      tdm_load_tile_f32(Bg + (size_t)k0 * N + n0, BgB, BN, BK,
                        (unsigned long long)N);
      tdm_load_tile_f32(Bu + (size_t)k0 * N + n0, BuB, BN, BK,
                        (unsigned long long)N);
    }
#endif
    #pragma unroll
    for (int i = 0; i < 2; ++i) {
      const int c = tid * 2 + i;
      const int row = c >> 2, sub = c & 3;
      const _Float16* g = A + (size_t)(m0 + row) * K + k0 + sub * 8;
      async_cp16(AsmB + (unsigned)(row * AROW + sub * 8) * 2, g);
    }
    async_wait0();
#if HAS_TDM
    if (w == 0) __builtin_amdgcn_s_wait_tensorcnt(0);
#else
    for (int id = tid; id < BK * BN; id += 256) {
      const int r2 = id >> 7, c2 = id & 127;
      Bfg[r2 * BROW + c2] = Bg[(size_t)(k0 + r2) * N + n0 + c2];
      Bfu[r2 * BROW + c2] = Bu[(size_t)(k0 + r2) * N + n0 + c2];
    }
#endif
    __syncthreads();

    {
      union { uint4 u[2]; _Float16 h[16]; } gf, uf;
      #pragma unroll
      for (int j = 0; j < 16; ++j) {
        gf.h[j] = (_Float16)Bfg[(bkh * 16 + j) * BROW + bcol];
        uf.h[j] = (_Float16)Bfu[(bkh * 16 + j) * BROW + bcol];
      }
      *(uint4*)&Gsm[boff]     = gf.u[0];
      *(uint4*)&Gsm[boff + 8] = gf.u[1];
      *(uint4*)&Usm[boff]     = uf.u[0];
      *(uint4*)&Usm[boff + 8] = uf.u[1];
    }
    __syncthreads();

    const int eo = hi ? 16 : 0;
    v16h af[2], gfr[4], ufr[4];
    #pragma unroll
    for (int i = 0; i < 2; ++i) {
      const _Float16* pr = &Asm[(wr * 32 + i * 16 + rloc) * AROW + eo];
      af[i] = ld_frag2(pr, pr + 8);
    }
    #pragma unroll
    for (int j = 0; j < 4; ++j) {
      gfr[j] = ld_frag(&Gsm[(wc * 4 + j) * 512 + lane * 16]);
      ufr[j] = ld_frag(&Usm[(wc * 4 + j) * 512 + lane * 16]);
    }
    #pragma unroll
    for (int i = 0; i < 2; ++i)
      #pragma unroll
      for (int j = 0; j < 4; ++j) {
        ag[i][j] = wmma32(af[i], gfr[j], ag[i][j]);
        au[i][j] = wmma32(af[i], ufr[j], au[i][j]);
      }
    __syncthreads();
  }

  const int rbase = m0 + wr * 32 + (hi ? 8 : 0);
  const int cbase = n0 + wc * 64 + rloc;
  #pragma unroll
  for (int i = 0; i < 2; ++i)
    #pragma unroll
    for (int j = 0; j < 4; ++j)
      #pragma unroll
      for (int r = 0; r < 8; ++r) {
        float g = ag[i][j][r];
        float u = au[i][j][r];
        float tt = 0.7978845608f * (g + 0.044715f * g * g * g);
        float gl = 0.5f * g * (1.0f + tanhf(tt));
        Ch[(size_t)(rbase + i * 16 + r) * N + (cbase + j * 16)] =
            (_Float16)(u * gl);
      }
}

/* ---------------- RoPE + kv_fused emit ---------------- */

__global__ __launch_bounds__(256)
void k_rope(const int* __restrict__ pos, _Float16* __restrict__ q,
            _Float16* __restrict__ kk, const _Float16* __restrict__ vv,
            float* __restrict__ kv_k, float* __restrict__ kv_v) {
  const int t = blockIdx.x;
  const float p = (float)pos[t];
  for (int idx = threadIdx.x; idx < (NH + NKV) * (DH / 2); idx += 256) {
    const int head = idx >> 7;
    const int d = idx & 127;
    const float inv = __powf(10000.0f, -(float)d * (1.0f / 128.0f));
    float sn, cs;
    __sincosf(p * inv, &sn, &cs);
    _Float16* base = (head < NH)
        ? (q + ((size_t)t * NH + head) * DH)
        : (kk + ((size_t)t * NKV + (head - NH)) * DH);
    const float x1 = (float)base[d], x2 = (float)base[d + 128];
    const float o1 = x1 * cs - x2 * sn;
    const float o2 = x2 * cs + x1 * sn;
    base[d] = (_Float16)o1;
    base[d + 128] = (_Float16)o2;
    if (head >= NH) {
      const size_t kb = ((size_t)t * NKV + (head - NH)) * DH;
      kv_k[kb + d] = o1;
      kv_k[kb + d + 128] = o2;
    }
  }
  for (int idx = threadIdx.x; idx < NKV * DH; idx += 256)
    kv_v[(size_t)t * NKV * DH + idx] = (float)vv[(size_t)t * NKV * DH + idx];
}

/* ---------------- WMMA sliding-window flash attention ---------------- */

__global__ __launch_bounds__(128)
void k_attn(const _Float16* __restrict__ q, const _Float16* __restrict__ k,
            const _Float16* __restrict__ v, _Float16* __restrict__ o) {
  __shared__ _Float16 Ksm[32 * KROW];  /* row-major padded, async-staged */
  __shared__ _Float16 Vsm[16 * 512];   /* frag-major                     */
  __shared__ _Float16 Psm[4 * 512];

  const int tid = threadIdx.x;
  const int lane = tid & 31;
  const int w = tid >> 5;
  const int rloc = lane & 15;
  const bool hi = lane >= 16;
  const int b = blockIdx.z;
  const int hq = blockIdx.y;
  const int nkv = hq >> 1;
  const int q0 = blockIdx.x * 64;
  const unsigned KsmB = lds_off32(&Ksm[0]);

  /* Q fragments: two contiguous 16B runs per fragment, straight from global */
  v16h Qf[8];
  {
    const _Float16* qrow =
        q + ((size_t)(b * SS + q0 + w * 16 + rloc) * NH + hq) * DH;
    const int eo = hi ? 16 : 0;
    #pragma unroll
    for (int kk = 0; kk < 8; ++kk) {
      const _Float16* pq = qrow + kk * 32 + eo;
      Qf[kk] = ld_frag2(pq, pq + 8);
    }
  }

  float m[8], l[8];
  v8f O[16];
  #pragma unroll
  for (int r = 0; r < 8; ++r) { m[r] = -1e30f; l[r] = 0.f; }
  #pragma unroll
  for (int n2 = 0; n2 < 16; ++n2) O[n2] = zero8();

  const int jmin = (q0 - (WINDOW - 1)) > 0 ? (q0 - (WINDOW - 1)) : 0;
  const int kb_lo = jmin & ~31;
  const int kb_hi = q0 + 63;

  for (int kb = kb_lo; kb <= kb_hi; kb += 32) {
    /* K tile: async copy, row-major padded LDS */
    for (int c = tid; c < 1024; c += 128) {
      const int row = c >> 5, sub = c & 31;
      const _Float16* g =
          k + ((size_t)(b * SS + kb + row) * NKV + nkv) * DH + sub * 8;
      async_cp16(KsmB + (unsigned)(row * KROW + sub * 8) * 2, g);
    }
    /* V tile: transpose-scatter into frag-major layout */
    for (int task = tid; task < 512; task += 128) {
      const int dh = task & 255;
      const int kh2 = task >> 8;
      union { uint4 u[2]; _Float16 h[16]; } f;
      #pragma unroll
      for (int j = 0; j < 16; ++j)
        f.h[j] = v[((size_t)(b * SS + kb + kh2 * 16 + j) * NKV + nkv) * DH + dh];
      const int n2 = dh >> 4;
      const int ln = (dh & 15) + (kh2 << 4);
      uint4* dst = (uint4*)&Vsm[n2 * 512 + ln * 16];
      dst[0] = f.u[0]; dst[1] = f.u[1];
    }
    async_wait0();
    __syncthreads();

    v8f sc[2];
    sc[0] = zero8(); sc[1] = zero8();
    #pragma unroll
    for (int nt = 0; nt < 2; ++nt)
      #pragma unroll
      for (int kk = 0; kk < 8; ++kk)
        sc[nt] = wmma32(Qf[kk],
                        ld_frag(&Ksm[(nt * 16 + rloc) * KROW + kk * 32 +
                                     (hi ? 16 : 0)]),
                        sc[nt]);

    #pragma unroll
    for (int r = 0; r < 8; ++r) {
      const int irow = q0 + w * 16 + r + (hi ? 8 : 0);
      float sv[2];
      float tmax = -1e30f;
      #pragma unroll
      for (int nt = 0; nt < 2; ++nt) {
        float s = sc[nt][r] * ATT_SCALE;
        s = LOGIT_CAP * tanhf(s * (1.0f / LOGIT_CAP));
        const int j = kb + nt * 16 + rloc;
        if (j > irow || (irow - j) >= WINDOW) s = -1e30f;
        sv[nt] = s;
        tmax = fmaxf(tmax, s);
      }
      #pragma unroll
      for (int off = 8; off; off >>= 1)
        tmax = fmaxf(tmax, __shfl_xor(tmax, off, 32));
      const float mn = fmaxf(m[r], tmax);
      const float alpha = __expf(m[r] - mn);
      const float p0 = (sv[0] < -1e29f) ? 0.f : __expf(sv[0] - mn);
      const float p1 = (sv[1] < -1e29f) ? 0.f : __expf(sv[1] - mn);
      float ps = p0 + p1;
      #pragma unroll
      for (int off = 8; off; off >>= 1) ps += __shfl_xor(ps, off, 32);
      l[r] = l[r] * alpha + ps;
      m[r] = mn;
      #pragma unroll
      for (int n2 = 0; n2 < 16; ++n2) O[n2][r] *= alpha;

      const int prow = r + (hi ? 8 : 0);
      {
        int kkey = rloc;
        int region = kkey >> 3;
        int pl = prow + ((region & 2) ? 16 : 0);
        int pe = kkey & 15;
        Psm[w * 512 + pl * 16 + pe] = (_Float16)p0;
        kkey = 16 + rloc;
        region = kkey >> 3;
        pl = prow + ((region & 2) ? 16 : 0);
        pe = kkey & 15;
        Psm[w * 512 + pl * 16 + pe] = (_Float16)p1;
      }
    }
    __syncthreads();

    v16h Pf = ld_frag(&Psm[w * 512 + lane * 16]);
    #pragma unroll
    for (int n2 = 0; n2 < 16; ++n2)
      O[n2] = wmma32(Pf, ld_frag(&Vsm[n2 * 512 + lane * 16]), O[n2]);
    __syncthreads();
  }

  #pragma unroll
  for (int r = 0; r < 8; ++r) {
    const float linv = 1.0f / l[r];
    const int srow = q0 + w * 16 + r + (hi ? 8 : 0);
    _Float16* orow = o + ((size_t)(b * SS + srow) * NH + hq) * DH;
    #pragma unroll
    for (int n2 = 0; n2 < 16; ++n2)
      orow[n2 * 16 + rloc] = (_Float16)(O[n2][r] * linv);
  }
}

/* ---------------- launch ---------------- */

extern "C" void kernel_launch(void* const* d_in, const int* in_sizes, int n_in,
                              void* d_out, int out_size, void* d_ws,
                              size_t ws_size, hipStream_t stream) {
  const float* hidden         = (const float*)d_in[0];
  const int*   positions      = (const int*)d_in[1];
  const float* in_ln_w        = (const float*)d_in[2];
  const float* post_attn_ln_w = (const float*)d_in[3];
  const float* pre_ff_ln_w    = (const float*)d_in[4];
  const float* post_ff_ln_w   = (const float*)d_in[5];
  const float* q_w    = (const float*)d_in[6];
  const float* k_w    = (const float*)d_in[7];
  const float* v_w    = (const float*)d_in[8];
  const float* o_w    = (const float*)d_in[9];
  const float* gate_w = (const float*)d_in[10];
  const float* up_w   = (const float*)d_in[11];
  const float* down_w = (const float*)d_in[12];

  float* out_x    = (float*)d_out;
  float* out_res  = out_x + (size_t)TT * HID;
  float* out_kv_k = out_res + (size_t)TT * HID;
  float* out_kv_v = out_kv_k + (size_t)TT * NKV * DH;

  char* p = (char*)d_ws;
  _Float16* xh = (_Float16*)p; p += (size_t)TT * HID * 2;
  _Float16* qh = (_Float16*)p; p += (size_t)TT * NH * DH * 2;
  _Float16* kh = (_Float16*)p; p += (size_t)TT * NKV * DH * 2;
  _Float16* vh = (_Float16*)p; p += (size_t)TT * NKV * DH * 2;
  _Float16* ah = (_Float16*)p; p += (size_t)TT * NH * DH * 2;
  float*    tf = (float*)p;    p += (size_t)TT * HID * 4;
  _Float16* ih = (_Float16*)p; p += (size_t)TT * INTER * 2;

  k_rms_f16<<<TT, 256, 0, stream>>>(hidden, in_ln_w, xh);

  k_gemm<<<dim3(TT / BM, (NH * DH) / BN), 256, 0, stream>>>(
      xh, q_w, qh, nullptr, NH * DH, HID);
  k_gemm<<<dim3(TT / BM, (NKV * DH) / BN), 256, 0, stream>>>(
      xh, k_w, kh, nullptr, NKV * DH, HID);
  k_gemm<<<dim3(TT / BM, (NKV * DH) / BN), 256, 0, stream>>>(
      xh, v_w, vh, nullptr, NKV * DH, HID);

  k_rope<<<TT, 256, 0, stream>>>(positions, qh, kh, vh, out_kv_k, out_kv_v);

  k_attn<<<dim3(SS / 64, NH, BB), 128, 0, stream>>>(qh, kh, vh, ah);

  k_gemm<<<dim3(TT / BM, HID / BN), 256, 0, stream>>>(
      ah, o_w, nullptr, tf, HID, NH * DH);

  k_rms_addres<<<TT, 256, 0, stream>>>(tf, post_attn_ln_w, hidden, out_res);
  k_rms_f16<<<TT, 256, 0, stream>>>(out_res, pre_ff_ln_w, xh);

  k_gateup<<<dim3(TT / BM, INTER / BN), 256, 0, stream>>>(
      xh, gate_w, up_w, ih, INTER, HID);

  k_gemm<<<dim3(TT / BM, HID / BN), 256, 0, stream>>>(
      ih, down_w, nullptr, tf, HID, INTER);

  k_rms_f32<<<TT, 256, 0, stream>>>(tf, post_ff_ln_w, out_x);
}
